// XrayNet_3985729650975
// MI455X (gfx1250) — matrix operand order repaired and checked by reading, not compiled, not run on hardware
//
#include <hip/hip_runtime.h>

#define DEV __device__ __forceinline__

typedef _Float16 half16 __attribute__((ext_vector_type(16)));
typedef float    float8 __attribute__((ext_vector_type(8)));

static constexpr int VOC  = 32000;
static constexpr int EDIM = 256;
static constexpr int HDIM = 512;
static constexpr int BSZ  = 32;
static constexpr int LSEQ = 128;
static constexpr int FOURH = 4 * HDIM;      // 2048
static constexpr int ROWS  = BSZ * LSEQ;    // 4096
static constexpr int NT_G  = FOURH / 16;    // 128  n-tiles over gates
static constexpr int KT_H  = HDIM / 32;     // 16   k-tiles over H (f16 k=32)
static constexpr int KT_E  = EDIM / 32;     // 8    k-tiles over E
static constexpr int MT_R  = ROWS / 16;     // 256  m-tiles over (t,b) rows
static constexpr int NWG_REC = 16;          // one WG per k-tile of H

// ---------------- WMMA helper (gfx1250: V_WMMA_F32_16X16X32_F16) -------------
DEV float8 wmma_f16(half16 a, half16 b, float8 c) {
  return __builtin_amdgcn_wmma_f32_16x16x32_f16(
      /*neg_a=*/false, a, /*neg_b=*/false, b,
      /*c_mod=*/(short)0, c, /*reuse_a=*/false, /*reuse_b=*/false);
}

// ---- A-matrix (16x32 f16) inverse layout: (m,kk) -> (lane, half-index) ------
// ISA 7.12.2: lanes 0-15 (h=0): V0..V3 K=0..7, V4..V7 K=16..23
//             lanes16-31 (h=1): V0..V3 K=8..15, V4..V7 K=24..31
DEV void invA(int m, int kk, int& lane, int& hh) {
  int h, r;
  if (kk < 16) { h = (kk >> 3) & 1; r = (kk & 7) >> 1; }
  else { int k2 = kk - 16; h = (k2 >> 3) & 1; r = 4 + ((k2 & 7) >> 1); }
  lane = m + (h << 4);
  hh = (r << 1) | (kk & 1);
}

DEV float sigmoidf(float x) { return 1.0f / (1.0f + expf(-x)); }

// ---------------- init: zero state frags, barrier, fused biases --------------
__global__ void k_init(_Float16* h1frag, _Float16* h2A, unsigned* bar,
                       float* bias12, float* bias2s,
                       const float* b_ih1, const float* b_hh1,
                       const float* b_ih2, const float* b_hh2) {
  int i = blockIdx.x * blockDim.x + threadIdx.x;
  int stride = gridDim.x * blockDim.x;
  for (int j = i; j < 2 * 2 * KT_H * 512; j += stride) h1frag[j] = (_Float16)0.0f;
  for (int j = i; j < 2 * KT_H * 512; j += stride) h2A[j] = (_Float16)0.0f; // slot 0
  for (int j = i; j < FOURH; j += stride) {
    bias12[j] = b_ih1[j] + b_hh1[j];
    bias2s[j] = b_ih2[j] + b_hh2[j];
  }
  if (i < 8) bar[i] = 0u;
}

// ---- swizzle fp32 weight (N rows x K cols, out = x @ W^T) into B-fragments --
// B tile (32k x 16n): lane l -> n = l&15, h = l>>4; half s -> k = 8h+s (s<8), 16+8h+(s-8) (s>=8)
__global__ void k_swizzleB(const float* __restrict__ W, _Float16* __restrict__ frag,
                           int N, int K) {
  long total = (long)N * K;
  long i = (long)blockIdx.x * blockDim.x + threadIdx.x;
  if (i >= total) return;
  int hh = (int)(i & 15);
  int lane = (int)((i >> 4) & 31);
  long tile = i >> 9;
  int KT = K >> 5;
  int kt = (int)(tile % KT);
  int nt = (int)(tile / KT);
  int n = nt * 16 + (lane & 15);
  int k = kt * 32 + ((hh >> 3) << 4) + ((lane >> 4) << 3) + (hh & 7);
  frag[i] = (_Float16)W[(long)n * K + k];
}

// ---- gather tokens -> embedding rows, write as WMMA A-fragments (f16) -------
__global__ void k_embA(const float* __restrict__ emb, const int* __restrict__ gt,
                       _Float16* __restrict__ embA) {
  int i = blockIdx.x * blockDim.x + threadIdx.x;   // ROWS * EDIM threads
  if (i >= ROWS * EDIM) return;
  int k = i & (EDIM - 1);
  int row = i >> 8;               // row = t*32 + b
  int t = row >> 5, b = row & 31;
  int tok = (t == 0) ? 0 : gt[b * LSEQ + (t - 1)];   // SOS = 0
  float v = emb[(long)tok * EDIM + k];
  int mt = row >> 4, m = row & 15, kt = k >> 5, kk = k & 31;
  int lane, hh; invA(m, kk, lane, hh);
  embA[(((long)(mt * KT_E + kt)) * 32 + lane) * 16 + hh] = (_Float16)v;
}

// ---- parallel precompute: xg1[t,b,:] = emb @ w_ih1^T + (b_ih1+b_hh1) --------
__global__ void __launch_bounds__(256) k_xgates(const _Float16* __restrict__ embA,
                                                const _Float16* __restrict__ wih1f,
                                                const float* __restrict__ bias12,
                                                float* __restrict__ xg1) {
  int wid = threadIdx.x >> 5, lane = threadIdx.x & 31;
  int job = blockIdx.x * 8 + wid;       // MT_R * NT_G = 32768 wave-jobs
  int nt = job & (NT_G - 1);
  int mt = job >> 7;
  const half16* A = (const half16*)embA;
  const half16* Bf = (const half16*)wih1f;
  float bv = bias12[nt * 16 + (lane & 15)];
  float8 acc;
  for (int r = 0; r < 8; ++r) acc[r] = bv;
#pragma unroll
  for (int kt = 0; kt < KT_E; ++kt) {
    half16 a = A[(mt * KT_E + kt) * 32 + lane];
    half16 b = Bf[(nt * KT_E + kt) * 32 + lane];
    acc = wmma_f16(a, b, acc);
  }
  int n = nt * 16 + (lane & 15);
  int mb = 8 * (lane >> 4);
#pragma unroll
  for (int r = 0; r < 8; ++r)
    xg1[(long)(mt * 16 + r + mb) * FOURH + n] = acc[r];
}

// ---------------- global sense-reversing barrier -----------------------------
DEV void gbarrier(unsigned* bar) {
  __syncthreads();
  if (threadIdx.x == 0) {
    __threadfence();
    volatile unsigned* vgen = bar + 1;
    unsigned g = *vgen;
    if (atomicAdd(bar, 1u) == NWG_REC - 1) {
      *bar = 0u;
      __threadfence();
      atomicAdd((unsigned*)(bar + 1), 1u);
    } else {
      while (*vgen == g) __builtin_amdgcn_s_sleep(2);
    }
    __threadfence();
  }
  __syncthreads();
}

// ---------------- persistent recurrent kernel (16 WGs, 2 barriers/step) ------
__global__ void __launch_bounds__(256) k_rec(
    const float* __restrict__ cnn, const float* __restrict__ xg1,
    const _Float16* __restrict__ whh1f, const _Float16* __restrict__ wih2f,
    const _Float16* __restrict__ whh2f, const float* __restrict__ bias2s,
    _Float16* h1frag, _Float16* h2A, unsigned* bar) {
  __shared__ float gbuf[32 * 128];      // 16 KB gate staging for this WG's slice
  const int w = blockIdx.x;             // owns k-tile w (k in [w*32, w*32+32))
  const int tid = threadIdx.x;
  const int wid = tid >> 5, lane = tid & 31;

  float c1r[4], c2r[4];
  for (int j = 0; j < 4; ++j) {
    int idx = tid + 256 * j; int b = idx & 31, kk = idx >> 5;
    c1r[j] = cnn[b * HDIM + w * 32 + kk];   // c1 init = cnn_output
    c2r[j] = 0.0f;
  }
  const half16* Bhh1 = (const half16*)whh1f;
  const half16* Bih2 = (const half16*)wih2f;
  const half16* Bhh2 = (const half16*)whh2f;
  const half16* H1 = (const half16*)h1frag;
  const half16* H2 = (const half16*)h2A;

  for (int t = 0; t < LSEQ; ++t) {
    const int p = (t + 1) & 1, q = t & 1;   // h1frag double-buffer slots
    // ---- GEMM1: g1 = xg1[t] + h1 @ w_hh1^T (only our gate columns) ----
    for (int jj = 0; jj < 2; ++jj) {
      int job = wid * 2 + jj;
      int mtl = job & 1, ntl = job >> 1;
      int gi = ntl >> 1, hn = ntl & 1;
      int ntg = gi * 32 + w * 2 + hn;       // global n-tile
      int n = ntg * 16 + (lane & 15);
      int mb = 8 * (lane >> 4);
      float8 acc;
#pragma unroll
      for (int r = 0; r < 8; ++r)
        acc[r] = xg1[(long)(t * 32 + mtl * 16 + r + mb) * FOURH + n];
#pragma unroll
      for (int kt = 0; kt < KT_H; ++kt) {
        half16 a = H1[((p * 2 + mtl) * KT_H + kt) * 32 + lane];
        half16 b = Bhh1[(ntg * KT_H + kt) * 32 + lane];
        acc = wmma_f16(a, b, acc);
      }
      int cl = ntl * 16 + (lane & 15);
#pragma unroll
      for (int r = 0; r < 8; ++r)
        gbuf[(mtl * 16 + r + mb) * 128 + cl] = acc[r];
    }
    __syncthreads();
    // ---- LSTM1 elementwise (WG-local columns), write h1 A-fragments ----
    for (int j = 0; j < 4; ++j) {
      int idx = tid + 256 * j; int b = idx & 31, kk = idx >> 5;
      float gi_ = gbuf[b * 128 +  0 + kk];
      float gf_ = gbuf[b * 128 + 32 + kk];
      float gg_ = gbuf[b * 128 + 64 + kk];
      float go_ = gbuf[b * 128 + 96 + kk];
      float c = sigmoidf(gf_) * c1r[j] + sigmoidf(gi_) * tanhf(gg_);
      c1r[j] = c;
      float h = sigmoidf(go_) * tanhf(c);
      int lane2, hh; invA(b & 15, kk, lane2, hh);
      int mtl = b >> 4;
      h1frag[(((q * 2 + mtl) * KT_H + w) * 32 + lane2) * 16 + hh] = (_Float16)h;
    }
    gbarrier(bar);
    // ---- GEMM2: g2 = bias2 + h1 @ w_ih2^T + h2 @ w_hh2^T ----
    for (int jj = 0; jj < 2; ++jj) {
      int job = wid * 2 + jj;
      int mtl = job & 1, ntl = job >> 1;
      int gi = ntl >> 1, hn = ntl & 1;
      int ntg = gi * 32 + w * 2 + hn;
      float bv = bias2s[ntg * 16 + (lane & 15)];
      float8 acc;
#pragma unroll
      for (int r = 0; r < 8; ++r) acc[r] = bv;
#pragma unroll
      for (int kt = 0; kt < KT_H; ++kt) {
        half16 a1 = H1[((q * 2 + mtl) * KT_H + kt) * 32 + lane];
        half16 b1 = Bih2[(ntg * KT_H + kt) * 32 + lane];
        acc = wmma_f16(a1, b1, acc);
        half16 a2 = H2[((t * 2 + mtl) * KT_H + kt) * 32 + lane];  // slot t = state(t-1)
        half16 b2 = Bhh2[(ntg * KT_H + kt) * 32 + lane];
        acc = wmma_f16(a2, b2, acc);
      }
      int cl = ntl * 16 + (lane & 15);
      int mb = 8 * (lane >> 4);
#pragma unroll
      for (int r = 0; r < 8; ++r)
        gbuf[(mtl * 16 + r + mb) * 128 + cl] = acc[r];
    }
    __syncthreads();
    // ---- LSTM2 elementwise; write h2 A-fragments into history slot t+1 ----
    for (int j = 0; j < 4; ++j) {
      int idx = tid + 256 * j; int b = idx & 31, kk = idx >> 5;
      float gi_ = gbuf[b * 128 +  0 + kk];
      float gf_ = gbuf[b * 128 + 32 + kk];
      float gg_ = gbuf[b * 128 + 64 + kk];
      float go_ = gbuf[b * 128 + 96 + kk];
      float c = sigmoidf(gf_) * c2r[j] + sigmoidf(gi_) * tanhf(gg_);
      c2r[j] = c;
      float h = sigmoidf(go_) * tanhf(c);
      int lane2, hh; invA(b & 15, kk, lane2, hh);
      int mtl = b >> 4;
      h2A[((((t + 1) * 2 + mtl) * KT_H + w) * 32 + lane2) * 16 + hh] = (_Float16)h;
    }
    gbarrier(bar);
  }
}

// ---- big projection GEMM: logits = h2_seq @ w_out^T + b_out -> d_out --------
// Register-blocked 4 m-tiles x 4 n-tiles per wave: 16 WMMAs per 8 fragment
// loads (~32 flop per L2 byte) so the 134-GFLOP GEMM is not L2-bandwidth bound.
__global__ void __launch_bounds__(256) k_proj(const _Float16* __restrict__ h2A,
                                              const _Float16* __restrict__ woutf,
                                              const float* __restrict__ b_out,
                                              float* __restrict__ out) {
  int wid = threadIdx.x >> 5, lane = threadIdx.x & 31;
  int job = blockIdx.x * 8 + wid;     // (256/4) mgroups * (2000/4) ngroups = 32000
  int mt0 = (job / 500) * 4;
  int nt0 = (job % 500) * 4;
  const half16* A = (const half16*)h2A;
  const half16* Bf = (const half16*)woutf;
  float8 acc[16];
#pragma unroll
  for (int u = 0; u < 4; ++u) {
    float bv = b_out[(nt0 + u) * 16 + (lane & 15)];
#pragma unroll
    for (int i = 0; i < 4; ++i)
#pragma unroll
      for (int r = 0; r < 8; ++r) acc[i * 4 + u][r] = bv;
  }
  for (int kt = 0; kt < KT_H; ++kt) {
    half16 a[4];
#pragma unroll
    for (int i = 0; i < 4; ++i)
      a[i] = A[((mt0 + i + 2) * KT_H + kt) * 32 + lane];  // row-tile -> h2 slot
#pragma unroll
    for (int u = 0; u < 4; ++u) {
      half16 b = Bf[((long)(nt0 + u) * KT_H + kt) * 32 + lane];
#pragma unroll
      for (int i = 0; i < 4; ++i)
        acc[i * 4 + u] = wmma_f16(a[i], b, acc[i * 4 + u]);
    }
  }
  int mb = 8 * (lane >> 4);
#pragma unroll
  for (int i = 0; i < 4; ++i) {
#pragma unroll
    for (int u = 0; u < 4; ++u) {
      int n = (nt0 + u) * 16 + (lane & 15);
#pragma unroll
      for (int r = 0; r < 8; ++r) {
        int row = (mt0 + i) * 16 + r + mb;        // row = t*32 + b
        int t = row >> 5, bb = row & 31;
        out[((long)bb * LSEQ + t) * VOC + n] = acc[i * 4 + u][r];
      }
    }
  }
}

// ---- in-place log-softmax per row of d_out ----------------------------------
__global__ void __launch_bounds__(256) k_lsm(float* __restrict__ out) {
  __shared__ float sm[256], ss[256];
  long base = (long)blockIdx.x * VOC;
  float m = -__builtin_inff(), s = 0.0f;
  for (int j = threadIdx.x; j < VOC; j += 256) {
    float x = out[base + j];
    float nm = fmaxf(m, x);
    s = s * expf(m - nm) + expf(x - nm);
    m = nm;
  }
  sm[threadIdx.x] = m; ss[threadIdx.x] = s;
  __syncthreads();
  for (int o = 128; o > 0; o >>= 1) {
    if (threadIdx.x < o) {
      float m2 = sm[threadIdx.x + o], s2 = ss[threadIdx.x + o];
      float nm = fmaxf(sm[threadIdx.x], m2);
      ss[threadIdx.x] = ss[threadIdx.x] * expf(sm[threadIdx.x] - nm) + s2 * expf(m2 - nm);
      sm[threadIdx.x] = nm;
    }
    __syncthreads();
  }
  float lse = sm[0] + logf(ss[0]);
  for (int j = threadIdx.x; j < VOC; j += 256)
    out[base + j] -= lse;
}

// -----------------------------------------------------------------------------
extern "C" void kernel_launch(void* const* d_in, const int* in_sizes, int n_in,
                              void* d_out, int out_size, void* d_ws, size_t ws_size,
                              hipStream_t stream) {
  const float* cnn   = (const float*)d_in[0];
  const float* emb   = (const float*)d_in[1];
  const float* w_ih1 = (const float*)d_in[2];
  const float* w_hh1 = (const float*)d_in[3];
  const float* b_ih1 = (const float*)d_in[4];
  const float* b_hh1 = (const float*)d_in[5];
  const float* w_ih2 = (const float*)d_in[6];
  const float* w_hh2 = (const float*)d_in[7];
  const float* b_ih2 = (const float*)d_in[8];
  const float* b_hh2 = (const float*)d_in[9];
  const float* w_out = (const float*)d_in[10];
  const float* b_out = (const float*)d_in[11];
  const int*   gt    = (const int*)d_in[12];
  float* outp = (float*)d_out;

  // workspace carve (256B aligned)
  char* p = (char*)d_ws;
  auto carve = [&](size_t bytes) -> char* {
    char* r = p; p += (bytes + 255) & ~(size_t)255; return r;
  };
  _Float16* wih1f = (_Float16*)carve((size_t)NT_G * KT_E * 512 * 2);
  _Float16* whh1f = (_Float16*)carve((size_t)NT_G * KT_H * 512 * 2);
  _Float16* wih2f = (_Float16*)carve((size_t)NT_G * KT_H * 512 * 2);
  _Float16* whh2f = (_Float16*)carve((size_t)NT_G * KT_H * 512 * 2);
  _Float16* woutf = (_Float16*)carve((size_t)(VOC / 16) * KT_H * 512 * 2);
  _Float16* embA  = (_Float16*)carve((size_t)MT_R * KT_E * 512 * 2);
  _Float16* h1fr  = (_Float16*)carve((size_t)2 * 2 * KT_H * 512 * 2);
  _Float16* h2A   = (_Float16*)carve((size_t)(LSEQ + 1) * 2 * KT_H * 512 * 2);
  float*    xg1   = (float*)carve((size_t)ROWS * FOURH * 4);
  float*    bias12= (float*)carve((size_t)FOURH * 4);
  float*    bias2s= (float*)carve((size_t)FOURH * 4);
  unsigned* bar   = (unsigned*)carve(256);

  auto nb = [](long n) { return (int)((n + 255) / 256); };

  k_init<<<64, 256, 0, stream>>>(h1fr, h2A, bar, bias12, bias2s,
                                 b_ih1, b_hh1, b_ih2, b_hh2);
  k_swizzleB<<<nb((long)FOURH * EDIM), 256, 0, stream>>>(w_ih1, wih1f, FOURH, EDIM);
  k_swizzleB<<<nb((long)FOURH * HDIM), 256, 0, stream>>>(w_hh1, whh1f, FOURH, HDIM);
  k_swizzleB<<<nb((long)FOURH * HDIM), 256, 0, stream>>>(w_ih2, wih2f, FOURH, HDIM);
  k_swizzleB<<<nb((long)FOURH * HDIM), 256, 0, stream>>>(w_hh2, whh2f, FOURH, HDIM);
  k_swizzleB<<<nb((long)VOC * HDIM), 256, 0, stream>>>(w_out, woutf, VOC, HDIM);
  k_embA<<<nb((long)ROWS * EDIM), 256, 0, stream>>>(emb, gt, embA);
  k_xgates<<<MT_R * NT_G / 8, 256, 0, stream>>>(embA, wih1f, bias12, xg1);
  k_rec<<<NWG_REC, 256, 0, stream>>>(cnn, xg1, whh1f, wih2f, whh2f, bias2s,
                                     h1fr, h2A, bar);
  k_proj<<<MT_R * 500 / 8 / 4, 256, 0, stream>>>(h2A, woutf, b_out, outp);
  k_lsm<<<ROWS, 256, 0, stream>>>(outp);
}
